// MultiHeadAttention_13846974562819
// MI455X (gfx1250) — matrix operand order repaired and checked by reading, not compiled, not run on hardware
//
#include <hip/hip_runtime.h>
#include <hip/hip_bf16.h>

// ---------------------------------------------------------------------------
// MHA block for gfx1250 (MI455X). f16 WMMA compute, f32 accumulate.
//   E=1024 H=16 HD=64 B=8 S=1024, softmax over BATCH axis (legacy quirk).
// Pipeline:
//   cvt -> GEMM(Q/K/V) -> GEMM(per-head Qh/Kh/Vh) -> transpose Vh ->
//   fused batch-softmax attention (WMMA) -> GEMM(Wlast)+residual -> LayerNorm
// GEMM: 64x64 block tile, B staged in LDS by the Tensor Data Mover
// (double-buffered, s_wait_tensorcnt), 4 WMMA per A-fragment; B-fragment
// LDS reads hoisted ahead of the WMMA chain so one dscnt wait covers all 4.
// ---------------------------------------------------------------------------

typedef _Float16 f16;
typedef _Float16 v16h __attribute__((ext_vector_type(16)));
typedef _Float16 v8h  __attribute__((ext_vector_type(8)));
typedef float    v8f  __attribute__((ext_vector_type(8)));
typedef unsigned int u32x4 __attribute__((ext_vector_type(4)));
typedef int          i32x4 __attribute__((ext_vector_type(4)));
typedef int          i32x8 __attribute__((ext_vector_type(8)));

#define E_DIM  1024
#define H_NUM  16
#define HD_DIM 64
#define B_NUM  8
#define S_LEN  1024
#define M_ROWS (B_NUM * S_LEN)   // 8192

#if defined(__AMDGCN__) && __has_builtin(__builtin_amdgcn_tensor_load_to_lds) && \
    __has_builtin(__builtin_amdgcn_s_wait_tensorcnt)
#define HAVE_TDM 1
#else
#define HAVE_TDM 0
#endif

__device__ __forceinline__ v8f wmma_f16(v16h a, v16h b, v8f c) {
  return __builtin_amdgcn_wmma_f32_16x16x32_f16(false, a, false, b, (short)0, c,
                                                false, false);
}

// A-matrix 16x32 f16 fragment (ISA 7.12.2): lane L(0-15) holds row M=L,
// K = 0..7 and 16..23; lane L+16 holds same row, K = 8..15 and 24..31.
__device__ __forceinline__ v16h make_a_frag(const f16* row, int k0) {
  v8h lo = *(const v8h*)(row + k0);
  v8h hi = *(const v8h*)(row + k0 + 16);
  v16h r;
#pragma unroll
  for (int i = 0; i < 8; ++i) { r[i] = lo[i]; r[8 + i] = hi[i]; }
  return r;
}

// B-matrix 32x16 f16 fragment: lane holds column n = lane&15, K = 0..7 and
// 16..23 (lanes 0-15) or 8..15 and 24..31 (lanes 16-31); W stored [N][K].
__device__ __forceinline__ v16h make_b_frag(const f16* p) {
  v8h lo = *(const v8h*)(p);
  v8h hi = *(const v8h*)(p + 16);
  v16h r;
#pragma unroll
  for (int i = 0; i < 8; ++i) { r[i] = lo[i]; r[8 + i] = hi[i]; }
  return r;
}

// ---------------------------------------------------------------------------
// Stage a 64(N-rows) x 32(K-halves) B tile into LDS.
// TDM path: 2D D# (data_size=4B): tile 16 units x 64 rows, row stride K/2
// units; LDS destination is contiguous [64][32] halves. Issued by wave 0 only
// (TDM ignores EXEC; one descriptor per workgroup per K-step).
__device__ __forceinline__ void stage_b_tile(const f16* Wk, int K, f16* lds,
                                             int tid) {
#if HAVE_TDM
  if ((tid >> 5) == 0) {
    const unsigned long long ga = (unsigned long long)(uintptr_t)Wk;
    const unsigned la = (unsigned)(uintptr_t)lds;  // low 32b = LDS offset
    u32x4 g0 = {};
    g0[0] = 1u;                                    // count=1, user mode
    g0[1] = la;                                    // lds_addr
    g0[2] = (unsigned)ga;                          // global_addr[31:0]
    g0[3] = (unsigned)((ga >> 32) & 0x01FFFFFFu)   // global_addr[56:32]
            | (2u << 30);                          // type=2 ("image")
    const unsigned td0 = 16;                       // tensor_dim0 (4B units)
    const unsigned td1 = 64;                       // tensor_dim1 (rows)
    const unsigned st0 = (unsigned)(K >> 1);       // row stride (4B units)
    i32x8 g1 = {};
    g1[0] = (int)(2u << 16);                       // data_size=2 -> 4 bytes
    g1[1] = (int)((td0 & 0xffffu) << 16);          // tensor_dim0[15:0]
    g1[2] = (int)(((td0 >> 16) & 0xffffu) | ((td1 & 0xffffu) << 16));
    g1[3] = (int)(((td1 >> 16) & 0xffffu) | (16u << 16));  // tile_dim0=16
    g1[4] = (int)(64u);                            // tile_dim1=64, tile_dim2=0
    g1[5] = (int)st0;                              // tensor_dim0_stride[31:0]
    g1[6] = 0;
    g1[7] = 0;
    i32x4 gz = {};
#if __clang_major__ >= 23
    i32x8 gz8 = {};
    __builtin_amdgcn_tensor_load_to_lds(g0, g1, gz, gz, gz8, 0);
#else
    __builtin_amdgcn_tensor_load_to_lds(g0, g1, gz, gz, 0);
#endif
  }
#else
  // Fallback: cooperative 128-thread staging (256 x 16B chunks).
  for (int c = tid; c < 256; c += 128) {
    const int row = c >> 2, part = c & 3;
    *(v8h*)&lds[row * 32 + part * 8] =
        *(const v8h*)(Wk + (size_t)row * K + part * 8);
  }
#endif
}

__device__ __forceinline__ void stage_b_wait(int tid) {
#if HAVE_TDM
  if ((tid >> 5) == 0) __builtin_amdgcn_s_wait_tensorcnt(0);
#endif
}

// ---------------------------------------------------------------------------
__global__ void cvt_f32_to_f16_kernel(const float* __restrict__ src,
                                      f16* __restrict__ dst, int n) {
  int i = blockIdx.x * blockDim.x + threadIdx.x;
  if (i < n) dst[i] = (f16)src[i];
}

// ---------------------------------------------------------------------------
// Y[M,N] = A[M,K] @ W[N,K]^T + bias[N]  (+ resid[M,N])
// Block: 4 waves -> 64(M)x64(N); wave w owns rows [w*16, w*16+16), all 64 N.
__global__ void __launch_bounds__(128)
gemm_wmma_kernel(const f16* __restrict__ A, const f16* __restrict__ W,
                 const float* __restrict__ bias, f16* __restrict__ Yh,
                 float* __restrict__ Yf, const float* __restrict__ resid,
                 int M, int N, int K) {
  __shared__ __align__(16) f16 Bs[2][64 * 32];  // 8 KB double buffer
  const int tid = threadIdx.x;
  const int wave = tid >> 5;
  const int lane = tid & 31;
  const int rs = lane & 15;
  const int hi = lane >> 4;
  const int blockN = blockIdx.x * 64;
  const int tileM = blockIdx.y * 64 + wave * 16;

  const f16* arow = A + (size_t)(tileM + rs) * K;
  const f16* wbase = W + (size_t)blockN * K;
  const int ak = hi ? 8 : 0;
  const int bk = hi ? 8 : 0;

  v8f acc[4];
#pragma unroll
  for (int nd = 0; nd < 4; ++nd) acc[nd] = (v8f){};

  stage_b_tile(wbase, K, &Bs[0][0], tid);
  stage_b_wait(tid);
  __syncthreads();

  for (int k = 0; k < K; k += 32) {
    const int buf = (k >> 5) & 1;
    if (k + 32 < K) stage_b_tile(wbase + k + 32, K, &Bs[buf ^ 1][0], tid);

    __builtin_prefetch(arow + k + 256, 0, 1);  // global_prefetch_b8
    v16h af = make_a_frag(arow + k, ak);
    // Hoist all B-fragment LDS reads so one dscnt wait covers the chain.
    v16h bf[4];
#pragma unroll
    for (int nd = 0; nd < 4; ++nd)
      bf[nd] = make_b_frag(&Bs[buf][(nd * 16 + rs) * 32 + bk]);
#pragma unroll
    for (int nd = 0; nd < 4; ++nd) acc[nd] = wmma_f16(af, bf[nd], acc[nd]);

    if (k + 32 < K) stage_b_wait(tid);
    __syncthreads();
  }

#pragma unroll
  for (int nd = 0; nd < 4; ++nd) {
    const int n = blockN + nd * 16 + rs;
    const float bv = bias ? bias[n] : 0.0f;
#pragma unroll
    for (int i = 0; i < 8; ++i) {
      const int m = tileM + i + hi * 8;
      size_t idx = (size_t)m * N + n;
      float v = acc[nd][i] + bv;
      if (resid) v += resid[idx];
      if (Yh) Yh[idx] = (f16)v;
      if (Yf) Yf[idx] = v;
    }
  }
}

// ---------------------------------------------------------------------------
// Vh [m = b*S+t][col = h*64+d]  ->  VhT [(h*B+b)*64+d][t]   (f16, LDS-tiled)
__global__ void __launch_bounds__(256)
transpose_vh_kernel(const f16* __restrict__ Vh, f16* __restrict__ VhT) {
  __shared__ f16 tile[32][33];
  const int hb = blockIdx.z;              // h*B + b
  const int h = hb / B_NUM, b = hb % B_NUM;
  const int t0 = blockIdx.x * 32;
  const int d0 = blockIdx.y * 32;
  const int tx = threadIdx.x & 31;
  const int ty = threadIdx.x >> 5;        // 0..7
#pragma unroll
  for (int r = 0; r < 4; ++r) {
    int t = t0 + ty + r * 8;
    tile[ty + r * 8][tx] =
        Vh[(size_t)(b * S_LEN + t) * E_DIM + h * HD_DIM + d0 + tx];
  }
  __syncthreads();
#pragma unroll
  for (int r = 0; r < 4; ++r) {
    int d = d0 + ty + r * 8;
    VhT[((size_t)hb * HD_DIM + d) * S_LEN + t0 + tx] = tile[tx][ty + r * 8];
  }
}

// ---------------------------------------------------------------------------
// Fused attention. One wave per (head, 16-row s-tile), all 8 batches resident.
// Softmax is over the batch axis => purely per-lane across the 8 batch
// accumulators (identical (s,t) lane mapping), no cross-lane reduction.
__global__ void __launch_bounds__(32, 1)
attn_fused_kernel(const f16* __restrict__ Qh, const f16* __restrict__ Kh,
                  const f16* __restrict__ VhT, f16* __restrict__ ctx) {
  __shared__ __align__(16) f16 lattn[B_NUM][16][32];  // 8 KB
  const int lane = threadIdx.x & 31;
  const int rs = lane & 15;
  const int hi = lane >> 4;
  const int h = blockIdx.y;
  const int s0 = blockIdx.x * 16;
  const float scale = 0.125f;  // 1/sqrt(64)
  const int ak = hi ? 8 : 0;
  const int bk = hi ? 8 : 0;

  v8f cacc[B_NUM][4];
#pragma unroll
  for (int b = 0; b < B_NUM; ++b)
#pragma unroll
    for (int nd = 0; nd < 4; ++nd) cacc[b][nd] = (v8f){};

  for (int t = 0; t < S_LEN; t += 32) {
    // ---- scores: S_b = Qh_b(16x64) @ Kh_b(32x64)^T, two 16-wide n-subtiles
    v8f sacc[B_NUM][2];
#pragma unroll
    for (int b = 0; b < B_NUM; ++b) {
      const f16* qrow = Qh + ((size_t)(b * S_LEN + s0 + rs) * E_DIM + h * HD_DIM);
      const f16* k0r  = Kh + ((size_t)(b * S_LEN + t + rs) * E_DIM + h * HD_DIM);
      const f16* k1r  = Kh + ((size_t)(b * S_LEN + t + 16 + rs) * E_DIM + h * HD_DIM);
      __builtin_prefetch(k0r + 32 * E_DIM, 0, 1);
      v8f sv0 = {}, sv1 = {};
#pragma unroll
      for (int kp = 0; kp < 2; ++kp) {
        v16h af = make_a_frag(qrow + kp * 32, ak);
        v16h b0 = make_b_frag(k0r + kp * 32 + bk);
        v16h b1 = make_b_frag(k1r + kp * 32 + bk);
        sv0 = wmma_f16(af, b0, sv0);
        sv1 = wmma_f16(af, b1, sv1);
      }
      sacc[b][0] = sv0;
      sacc[b][1] = sv1;
    }

    // ---- softmax over the 8 batches (per lane, per element)
#pragma unroll
    for (int u = 0; u < 2; ++u) {
#pragma unroll
      for (int i = 0; i < 8; ++i) {
        float mx = -3.0e30f;
#pragma unroll
        for (int b = 0; b < B_NUM; ++b) mx = fmaxf(mx, sacc[b][u][i]);
        float e[B_NUM];
        float sum = 0.0f;
#pragma unroll
        for (int b = 0; b < B_NUM; ++b) {
          e[b] = __expf(scale * (sacc[b][u][i] - mx));
          sum += e[b];
        }
        const float inv = 1.0f / sum;
        const int sl = i + hi * 8;        // C-layout row
        const int tl = u * 16 + rs;       // C-layout col
#pragma unroll
        for (int b = 0; b < B_NUM; ++b)
          lattn[b][sl][tl] = (f16)(e[b] * inv);
      }
    }
    __syncthreads();

    // ---- ctx_b(16x64) += attn_b(16x32) @ Vh_b(32x64)
#pragma unroll
    for (int b = 0; b < B_NUM; ++b) {
      v16h af = make_a_frag(&lattn[b][rs][0], ak);
      v16h bf[4];
#pragma unroll
      for (int nd = 0; nd < 4; ++nd) {
        const f16* vr = VhT +
            ((size_t)((h * B_NUM + b) * HD_DIM + nd * 16 + rs) * S_LEN + t + bk);
        bf[nd] = make_b_frag(vr);
      }
#pragma unroll
      for (int nd = 0; nd < 4; ++nd)
        cacc[b][nd] = wmma_f16(af, bf[nd], cacc[b][nd]);
    }
    __syncthreads();
  }

  // ---- write ctx [b*S+s][h*64+d] (f16)
#pragma unroll
  for (int b = 0; b < B_NUM; ++b) {
#pragma unroll
    for (int nd = 0; nd < 4; ++nd) {
      const int col = h * HD_DIM + nd * 16 + rs;
#pragma unroll
      for (int i = 0; i < 8; ++i) {
        const int m = b * S_LEN + s0 + i + hi * 8;
        ctx[(size_t)m * E_DIM + col] = (f16)cacc[b][nd][i];
      }
    }
  }
}

// ---------------------------------------------------------------------------
__global__ void __launch_bounds__(256)
layernorm_kernel(const float* __restrict__ pre, const float* __restrict__ gamma,
                 const float* __restrict__ beta, float* __restrict__ out) {
  __shared__ float red[256];
  const int row = blockIdx.x;
  const int tid = threadIdx.x;
  const float* p = pre + (size_t)row * E_DIM;

  float s = 0.0f;
  for (int i = tid; i < E_DIM; i += 256) s += p[i];
  red[tid] = s;
  __syncthreads();
  for (int o = 128; o > 0; o >>= 1) {
    if (tid < o) red[tid] += red[tid + o];
    __syncthreads();
  }
  const float mu = red[0] * (1.0f / E_DIM);
  __syncthreads();

  float v = 0.0f;
  for (int i = tid; i < E_DIM; i += 256) {
    float d = p[i] - mu;
    v += d * d;
  }
  red[tid] = v;
  __syncthreads();
  for (int o = 128; o > 0; o >>= 1) {
    if (tid < o) red[tid] += red[tid + o];
    __syncthreads();
  }
  const float inv = rsqrtf(red[0] * (1.0f / E_DIM) + 1e-5f);

  for (int i = tid; i < E_DIM; i += 256)
    out[(size_t)row * E_DIM + i] = (p[i] - mu) * inv * gamma[i] + beta[i];
}

// ---------------------------------------------------------------------------
extern "C" void kernel_launch(void* const* d_in, const int* in_sizes, int n_in,
                              void* d_out, int out_size, void* d_ws,
                              size_t ws_size, hipStream_t stream) {
  const float* x     = (const float*)d_in[0];
  const float* WQ    = (const float*)d_in[1];
  const float* bQ    = (const float*)d_in[2];
  const float* WK    = (const float*)d_in[3];
  const float* bK    = (const float*)d_in[4];
  const float* WV    = (const float*)d_in[5];
  const float* bV    = (const float*)d_in[6];
  const float* hWQ   = (const float*)d_in[7];
  const float* hbQ   = (const float*)d_in[8];
  const float* hWK   = (const float*)d_in[9];
  const float* hbK   = (const float*)d_in[10];
  const float* hWV   = (const float*)d_in[11];
  const float* hbV   = (const float*)d_in[12];
  const float* Wlast = (const float*)d_in[13];
  const float* blast = (const float*)d_in[14];
  const float* gamma = (const float*)d_in[15];
  const float* beta  = (const float*)d_in[16];
  float* out = (float*)d_out;

  const size_t MH = (size_t)M_ROWS * E_DIM;  // 8M elements
  const size_t WH = (size_t)E_DIM * E_DIM;   // 1M elements

  char* ws = (char*)d_ws;
  f16* xh     = (f16*)ws;            ws += MH * 2;
  f16* WQh    = (f16*)ws;            ws += WH * 2;
  f16* WKh    = (f16*)ws;            ws += WH * 2;
  f16* WVh    = (f16*)ws;            ws += WH * 2;
  f16* hWQh   = (f16*)ws;            ws += WH * 2;
  f16* hWKh   = (f16*)ws;            ws += WH * 2;
  f16* hWVh   = (f16*)ws;            ws += WH * 2;
  f16* Wlasth = (f16*)ws;            ws += WH * 2;
  f16* Qf     = (f16*)ws;            ws += MH * 2;
  f16* Kf     = (f16*)ws;            ws += MH * 2;
  f16* Vf     = (f16*)ws;            ws += MH * 2;
  f16* Qhh    = (f16*)ws;            ws += MH * 2;
  f16* Khh    = (f16*)ws;            ws += MH * 2;
  f16* Vhh    = (f16*)ws;            ws += MH * 2;
  f16* VhT    = (f16*)ws;            ws += MH * 2;
  f16* ctx    = (f16*)ws;            ws += MH * 2;
  float* pre  = (float*)ws;          ws += MH * 4;

  // 1) fp32 -> f16 conversions
  cvt_f32_to_f16_kernel<<<(int)((MH + 255) / 256), 256, 0, stream>>>(x, xh, (int)MH);
  cvt_f32_to_f16_kernel<<<(int)((WH + 255) / 256), 256, 0, stream>>>(WQ, WQh, (int)WH);
  cvt_f32_to_f16_kernel<<<(int)((WH + 255) / 256), 256, 0, stream>>>(WK, WKh, (int)WH);
  cvt_f32_to_f16_kernel<<<(int)((WH + 255) / 256), 256, 0, stream>>>(WV, WVh, (int)WH);
  cvt_f32_to_f16_kernel<<<(int)((WH + 255) / 256), 256, 0, stream>>>(hWQ, hWQh, (int)WH);
  cvt_f32_to_f16_kernel<<<(int)((WH + 255) / 256), 256, 0, stream>>>(hWK, hWKh, (int)WH);
  cvt_f32_to_f16_kernel<<<(int)((WH + 255) / 256), 256, 0, stream>>>(hWV, hWVh, (int)WH);
  cvt_f32_to_f16_kernel<<<(int)((WH + 255) / 256), 256, 0, stream>>>(Wlast, Wlasth, (int)WH);

  // 2) Q/K/V full-width projections
  dim3 ggrid(E_DIM / 64, M_ROWS / 64);  // (16, 128), 4 waves per block
  gemm_wmma_kernel<<<ggrid, 128, 0, stream>>>(xh, WQh, bQ, Qf, nullptr, nullptr,
                                              M_ROWS, E_DIM, E_DIM);
  gemm_wmma_kernel<<<ggrid, 128, 0, stream>>>(xh, WKh, bK, Kf, nullptr, nullptr,
                                              M_ROWS, E_DIM, E_DIM);
  gemm_wmma_kernel<<<ggrid, 128, 0, stream>>>(xh, WVh, bV, Vf, nullptr, nullptr,
                                              M_ROWS, E_DIM, E_DIM);

  // 3) per-head projections (flattened: out col = h*64+d)
  gemm_wmma_kernel<<<ggrid, 128, 0, stream>>>(Qf, hWQh, hbQ, Qhh, nullptr,
                                              nullptr, M_ROWS, E_DIM, E_DIM);
  gemm_wmma_kernel<<<ggrid, 128, 0, stream>>>(Kf, hWKh, hbK, Khh, nullptr,
                                              nullptr, M_ROWS, E_DIM, E_DIM);
  gemm_wmma_kernel<<<ggrid, 128, 0, stream>>>(Vf, hWVh, hbV, Vhh, nullptr,
                                              nullptr, M_ROWS, E_DIM, E_DIM);

  // 4) transpose Vh for contiguous attn@V B-fragments
  transpose_vh_kernel<<<dim3(S_LEN / 32, HD_DIM / 32, H_NUM * B_NUM), 256, 0,
                        stream>>>(Vhh, VhT);

  // 5) fused batch-softmax attention
  attn_fused_kernel<<<dim3(S_LEN / 16, H_NUM), 32, 0, stream>>>(Qhh, Khh, VhT,
                                                                ctx);

  // 6) output projection + bias + residual (f32 out)
  gemm_wmma_kernel<<<ggrid, 128, 0, stream>>>(ctx, Wlasth, blast, nullptr, pre,
                                              x, M_ROWS, E_DIM, E_DIM);

  // 7) LayerNorm
  layernorm_kernel<<<M_ROWS, 256, 0, stream>>>(pre, gamma, beta, out);
}
